// MultiHeadAttention_2044404433628
// MI455X (gfx1250) — compile-verified, matrix operand order checked by reading, not run on hardware
//
#include <hip/hip_runtime.h>
#include <hip/hip_bf16.h>

// ---------------------------------------------------------------------------
// MHA for B=2, S=2048, D=768, H=12, dk=64 on gfx1250 (CDNA5, wave32, WMMA).
// All GEMMs use v_wmma_f32_16x16x32_bf16 (bf16 in, f32 accumulate).
// ---------------------------------------------------------------------------

#define D_MODEL 768
#define SEQ     2048
#define BATCH   2
#define HEADS   12
#define DK      64
#define MTOT    (BATCH * SEQ)      // 4096

typedef __attribute__((ext_vector_type(16))) __bf16 v16bf;
typedef __attribute__((ext_vector_type(8)))  float  v8f;

union FragB {
    v16bf v;
    uint4 q[2];
};

static __device__ __forceinline__ unsigned short f2bfu(float x) {
    union { __bf16 h; unsigned short u; } c;
    c.h = (__bf16)x;
    return c.u;
}

static __device__ __forceinline__ v8f wmma_bf16(v16bf a, v16bf b, v8f c) {
    // (neg_a, A, neg_b, B, c_mod, C, reuse_a, reuse_b)
    return __builtin_amdgcn_wmma_f32_16x16x32_bf16(false, a, false, b,
                                                   (short)0, c, false, false);
}

// ---------------------------------------------------------------------------
// Weight transpose + f32->bf16 convert:  Wt[n][k] = bf16(W[k][n])   (768x768)
// ---------------------------------------------------------------------------
__global__ __launch_bounds__(256) void transpose_cvt(const float* __restrict__ W,
                                                     unsigned short* __restrict__ Wt) {
    __shared__ float t[32][33];
    const int tx = threadIdx.x, ty = threadIdx.y;
    const int x  = blockIdx.x * 32 + tx;   // n
    const int y0 = blockIdx.y * 32;        // k base
#pragma unroll
    for (int i = 0; i < 4; ++i) {
        t[ty + i * 8][tx] = W[(y0 + ty + i * 8) * D_MODEL + x];
    }
    __syncthreads();
#pragma unroll
    for (int i = 0; i < 4; ++i) {
        const int n = blockIdx.x * 32 + ty + i * 8;
        const int k = y0 + tx;
        Wt[n * D_MODEL + k] = f2bfu(t[tx][ty + i * 8]);
    }
}

// ---------------------------------------------------------------------------
// Generic projection GEMM:  out[m,n] = sum_k A[m,k] * Wt[n,k] + bias[n]
//   A: [4096 x 768] f32 (ABF16=false) or bf16 (ABF16=true), row-major
//   Wt: bf16 [768 x 768] (pre-transposed weight, contiguous in k)
//   MODE 0: out bf16 [B,H,S,DK]          (Q, K)
//   MODE 1: out bf16 [B,H,DK,S]          (V transposed)
//   MODE 2: out f32  [m,n] = [B*S, 768]  (final projection -> d_out)
// Block: 256 threads = 8 waves, tile 128(M) x 128(N); wave tile 32x64.
// ---------------------------------------------------------------------------
template <int MODE, bool ABF16>
__global__ __launch_bounds__(256) void gemm_proj(const void* __restrict__ A_,
                                                 const unsigned short* __restrict__ Wt,
                                                 const float* __restrict__ bias,
                                                 void* __restrict__ out_) {
    const int lane  = threadIdx.x & 31;
    const int wid   = threadIdx.x >> 5;
    const int lr    = lane & 15;
    const int hi16  = lane >> 4;
    const int kbase = hi16 << 3;       // 0 or 8
    const int mw    = wid & 3;
    const int nw    = wid >> 2;
    const int m0    = blockIdx.x * 128 + mw * 32;
    const int n0    = blockIdx.y * 128 + nw * 64;

    v8f acc[2][4] = {};

    for (int kk = 0; kk < D_MODEL; kk += 32) {
        FragB a[2];
        if (ABF16) {
            const unsigned short* Ab = (const unsigned short*)A_;
#pragma unroll
            for (int mt = 0; mt < 2; ++mt) {
                const unsigned short* p = Ab + (m0 + mt * 16 + lr) * D_MODEL + kk + kbase;
                a[mt].q[0] = *(const uint4*)(p);
                a[mt].q[1] = *(const uint4*)(p + 16);
            }
        } else {
            const float* Af = (const float*)A_;
#pragma unroll
            for (int mt = 0; mt < 2; ++mt) {
                const float* p = Af + (m0 + mt * 16 + lr) * D_MODEL + kk + kbase;
                float4 x0 = *(const float4*)(p);
                float4 x1 = *(const float4*)(p + 4);
                float4 y0 = *(const float4*)(p + 16);
                float4 y1 = *(const float4*)(p + 20);
                float h0[8] = {x0.x, x0.y, x0.z, x0.w, x1.x, x1.y, x1.z, x1.w};
                float h1[8] = {y0.x, y0.y, y0.z, y0.w, y1.x, y1.y, y1.z, y1.w};
#pragma unroll
                for (int i = 0; i < 8; ++i) {
                    a[mt].v[i]     = (__bf16)h0[i];
                    a[mt].v[8 + i] = (__bf16)h1[i];
                }
            }
        }

        FragB bfr[4];
#pragma unroll
        for (int nt = 0; nt < 4; ++nt) {
            const unsigned short* p = Wt + (n0 + nt * 16 + lr) * D_MODEL + kk + kbase;
            bfr[nt].q[0] = *(const uint4*)(p);
            bfr[nt].q[1] = *(const uint4*)(p + 16);
        }

#pragma unroll
        for (int mt = 0; mt < 2; ++mt)
#pragma unroll
            for (int nt = 0; nt < 4; ++nt)
                acc[mt][nt] = wmma_bf16(a[mt].v, bfr[nt].v, acc[mt][nt]);
    }

    // Epilogue
    float bv[4];
#pragma unroll
    for (int nt = 0; nt < 4; ++nt) bv[nt] = bias[n0 + nt * 16 + lr];

#pragma unroll
    for (int mt = 0; mt < 2; ++mt) {
#pragma unroll
        for (int nt = 0; nt < 4; ++nt) {
            const int col = n0 + nt * 16 + lr;
#pragma unroll
            for (int r = 0; r < 8; ++r) {
                const int row = m0 + mt * 16 + r + hi16 * 8;
                const float val = acc[mt][nt][r] + bv[nt];
                if (MODE == 2) {
                    ((float*)out_)[row * D_MODEL + col] = val;
                } else {
                    const int b = row >> 11;          // /SEQ
                    const int s = row & (SEQ - 1);
                    const int h = col >> 6;           // /DK
                    const int d = col & (DK - 1);
                    unsigned short* ob = (unsigned short*)out_;
                    if (MODE == 0)
                        ob[((b * HEADS + h) * SEQ + s) * DK + d] = f2bfu(val);
                    else // MODE 1: V^T
                        ob[((b * HEADS + h) * DK + d) * SEQ + s] = f2bfu(val);
                }
            }
        }
    }
}

// ---------------------------------------------------------------------------
// Flash attention: one wave handles 16 query rows; keys streamed in chunks
// of 32 with online softmax.  Q,K bf16 [B,H,S,64]; Vt bf16 [B,H,64,S].
// Output bf16 [B,S,H*64] (heads re-concatenated) for the final projection.
// ---------------------------------------------------------------------------
__global__ __launch_bounds__(256) void flash_attn(const unsigned short* __restrict__ Q,
                                                  const unsigned short* __restrict__ K,
                                                  const unsigned short* __restrict__ Vt,
                                                  const int* __restrict__ mask,
                                                  unsigned short* __restrict__ AO) {
    __shared__ __align__(16) unsigned short Plds[8][16][32];

    const int lane  = threadIdx.x & 31;
    const int wid   = threadIdx.x >> 5;
    const int lr    = lane & 15;
    const int hi16  = lane >> 4;
    const int kbase = hi16 << 3;
    const int b = blockIdx.z;
    const int h = blockIdx.y;
    const int q0 = blockIdx.x * 128 + wid * 16;

    const unsigned short* Qh = Q  + ((b * HEADS + h) * SEQ) * DK;
    const unsigned short* Kh = K  + ((b * HEADS + h) * SEQ) * DK;
    const unsigned short* Vh = Vt + ((b * HEADS + h) * DK) * SEQ;
    const int* mk = mask + b * SEQ;

    // Q fragments (persist over the key loop): dk halves 0..31 and 32..63
    FragB qf[2];
#pragma unroll
    for (int dh = 0; dh < 2; ++dh) {
        const unsigned short* p = Qh + (q0 + lr) * DK + dh * 32 + kbase;
        qf[dh].q[0] = *(const uint4*)(p);
        qf[dh].q[1] = *(const uint4*)(p + 16);
    }

    v8f o[4] = {};
    float mrow[8], lrow[8], alpha[8];
#pragma unroll
    for (int r = 0; r < 8; ++r) { mrow[r] = -1e30f; lrow[r] = 0.f; }

    for (int j = 0; j < SEQ; j += 32) {
        // --- S = Q * K^T (two 16-key tiles) ---
        FragB kf[2][2];
#pragma unroll
        for (int nt = 0; nt < 2; ++nt)
#pragma unroll
            for (int dh = 0; dh < 2; ++dh) {
                const unsigned short* p = Kh + (j + nt * 16 + lr) * DK + dh * 32 + kbase;
                kf[nt][dh].q[0] = *(const uint4*)(p);
                kf[nt][dh].q[1] = *(const uint4*)(p + 16);
            }
        v8f s0 = {}, s1 = {};
        s0 = wmma_bf16(qf[0].v, kf[0][0].v, s0);
        s0 = wmma_bf16(qf[1].v, kf[0][1].v, s0);
        s1 = wmma_bf16(qf[0].v, kf[1][0].v, s1);
        s1 = wmma_bf16(qf[1].v, kf[1][1].v, s1);

        // --- scale + mask ---
        const int mv0 = mk[j + lr];
        const int mv1 = mk[j + 16 + lr];
        float a0[8], a1[8];
#pragma unroll
        for (int r = 0; r < 8; ++r) {
            a0[r] = mv0 ? s0[r] * 0.125f : -1e9f;   // 1/sqrt(64)
            a1[r] = mv1 ? s1[r] * 0.125f : -1e9f;
        }

        // --- online softmax (row stats across 16-lane N groups) ---
#pragma unroll
        for (int r = 0; r < 8; ++r) {
            float c = fmaxf(a0[r], a1[r]);
            c = fmaxf(c, __shfl_xor(c, 8, 16));
            c = fmaxf(c, __shfl_xor(c, 4, 16));
            c = fmaxf(c, __shfl_xor(c, 2, 16));
            c = fmaxf(c, __shfl_xor(c, 1, 16));
            const float mn = fmaxf(mrow[r], c);
            alpha[r] = __expf(mrow[r] - mn);
            mrow[r]  = mn;
        }
#pragma unroll
        for (int r = 0; r < 8; ++r) {
            a0[r] = __expf(a0[r] - mrow[r]);
            a1[r] = __expf(a1[r] - mrow[r]);
            float s = a0[r] + a1[r];
            s += __shfl_xor(s, 8, 16);
            s += __shfl_xor(s, 4, 16);
            s += __shfl_xor(s, 2, 16);
            s += __shfl_xor(s, 1, 16);
            lrow[r] = lrow[r] * alpha[r] + s;
        }
#pragma unroll
        for (int t = 0; t < 4; ++t)
#pragma unroll
            for (int r = 0; r < 8; ++r) o[t][r] *= alpha[r];

        // --- transpose P (C layout -> A layout) through per-wave LDS tile ---
#pragma unroll
        for (int r = 0; r < 8; ++r) {
            const int row = r + hi16 * 8;
            Plds[wid][row][lr]      = f2bfu(a0[r]);
            Plds[wid][row][16 + lr] = f2bfu(a1[r]);
        }
        asm volatile("s_wait_dscnt 0" ::: "memory");  // DS in-order per wave
        FragB pf;
        pf.q[0] = *(const uint4*)&Plds[wid][lr][kbase];
        pf.q[1] = *(const uint4*)&Plds[wid][lr][kbase + 16];

        // --- O += P * V (V^T rows are contiguous in key dim) ---
#pragma unroll
        for (int dt = 0; dt < 4; ++dt) {
            FragB vf;
            const unsigned short* p = Vh + (dt * 16 + lr) * SEQ + j + kbase;
            vf.q[0] = *(const uint4*)(p);
            vf.q[1] = *(const uint4*)(p + 16);
            o[dt] = wmma_bf16(pf.v, vf.v, o[dt]);
        }
    }

    // --- normalize and store concat-head output bf16 [B, S, 768] ---
    float inv[8];
#pragma unroll
    for (int r = 0; r < 8; ++r) inv[r] = 1.0f / lrow[r];
#pragma unroll
    for (int dt = 0; dt < 4; ++dt) {
#pragma unroll
        for (int r = 0; r < 8; ++r) {
            const int s   = q0 + r + hi16 * 8;
            const int col = h * DK + dt * 16 + lr;
            AO[(b * SEQ + s) * D_MODEL + col] = f2bfu(o[dt][r] * inv[r]);
        }
    }
}

// ---------------------------------------------------------------------------
// Host-side orchestration
// ---------------------------------------------------------------------------
extern "C" void kernel_launch(void* const* d_in, const int* in_sizes, int n_in,
                              void* d_out, int out_size, void* d_ws, size_t ws_size,
                              hipStream_t stream) {
    (void)in_sizes; (void)n_in; (void)out_size; (void)ws_size;

    const float* query = (const float*)d_in[0];
    const float* keyi  = (const float*)d_in[1];
    const float* value = (const float*)d_in[2];
    const int*   mask  = (const int*)  d_in[3];
    const float* Wq    = (const float*)d_in[4];
    const float* bq    = (const float*)d_in[5];
    const float* Wk    = (const float*)d_in[6];
    const float* bk    = (const float*)d_in[7];
    const float* Wv    = (const float*)d_in[8];
    const float* bv    = (const float*)d_in[9];
    const float* Wo    = (const float*)d_in[10];
    const float* bo    = (const float*)d_in[11];

    char* ws = (char*)d_ws;
    const size_t WSZ  = (size_t)D_MODEL * D_MODEL * sizeof(unsigned short); // 1.18 MB
    const size_t QKSZ = (size_t)MTOT * D_MODEL * sizeof(unsigned short);    // 6.29 MB
    unsigned short* Wqt = (unsigned short*)(ws);
    unsigned short* Wkt = (unsigned short*)(ws + WSZ);
    unsigned short* Wvt = (unsigned short*)(ws + 2 * WSZ);
    unsigned short* Wot = (unsigned short*)(ws + 3 * WSZ);
    unsigned short* Qb  = (unsigned short*)(ws + 4 * WSZ);
    unsigned short* Kb  = (unsigned short*)(ws + 4 * WSZ + QKSZ);
    unsigned short* Vtb = (unsigned short*)(ws + 4 * WSZ + 2 * QKSZ);
    unsigned short* AO  = (unsigned short*)(ws + 4 * WSZ + 3 * QKSZ);

    // 1) weight transpose + convert
    dim3 tb(32, 8), tg(D_MODEL / 32, D_MODEL / 32);
    transpose_cvt<<<tg, tb, 0, stream>>>(Wq, Wqt);
    transpose_cvt<<<tg, tb, 0, stream>>>(Wk, Wkt);
    transpose_cvt<<<tg, tb, 0, stream>>>(Wv, Wvt);
    transpose_cvt<<<tg, tb, 0, stream>>>(Wo, Wot);

    // 2) Q/K/V projections (WMMA)
    dim3 gg(MTOT / 128, D_MODEL / 128);
    gemm_proj<0, false><<<gg, 256, 0, stream>>>(query, Wqt, bq, Qb);
    gemm_proj<0, false><<<gg, 256, 0, stream>>>(keyi,  Wkt, bk, Kb);
    gemm_proj<1, false><<<gg, 256, 0, stream>>>(value, Wvt, bv, Vtb);

    // 3) flash attention (WMMA, online softmax)
    flash_attn<<<dim3(SEQ / 128, HEADS, BATCH), 256, 0, stream>>>(Qb, Kb, Vtb, mask, AO);

    // 4) output projection -> f32 d_out
    gemm_proj<2, true><<<gg, 256, 0, stream>>>(AO, Wot, bo, d_out);
}
